// self_LSTM_sparse_embed_90907277787369
// MI455X (gfx1250) — matrix operand order repaired and checked by reading, not compiled, not run on hardware
//
#include <hip/hip_runtime.h>
#include <math.h>

// ---------------------------------------------------------------------------
// Types for CDNA5 WMMA (gfx1250, wave32)
// ---------------------------------------------------------------------------
typedef __attribute__((ext_vector_type(16))) __bf16 v16bf;
typedef __attribute__((ext_vector_type(8)))  float  v8f;

union Frag16 {
    v16bf v;
    uint4 q[2];
};

__device__ __forceinline__ unsigned short f32_to_bf16_rne(float f) {
    union { float f; unsigned int u; } c; c.f = f;
    unsigned int u = c.u;
    unsigned int r = u + 0x7FFFu + ((u >> 16) & 1u);
    if ((u & 0x7F800000u) == 0x7F800000u) r = u;   // Inf/NaN passthrough
    return (unsigned short)(r >> 16);
}

__device__ __forceinline__ v8f zero8() {
    v8f z = {0.f, 0.f, 0.f, 0.f, 0.f, 0.f, 0.f, 0.f};
    return z;
}

__device__ __forceinline__ v8f wmma_bf16(v16bf a, v16bf b, v8f c) {
    // D = A(16x32 bf16) x B(32x16 bf16) + C(16x16 f32)
    return __builtin_amdgcn_wmma_f32_16x16x32_bf16(
        /*neg_a=*/false, a, /*neg_b=*/false, b,
        /*c_mod=*/(short)0, c, /*reuse_a=*/false, /*reuse_b=*/false);
}

// A-matrix fragment: row-major source, per ISA 16-bit A 16x32 layout.
// lane<16:  elems 0..7 -> K = kt+0..7,   elems 8..15 -> K = kt+16..23
// lane>=16: elems 0..7 -> K = kt+8..15,  elems 8..15 -> K = kt+24..31
__device__ __forceinline__ v16bf load_a_frag(const unsigned short* rowPtr,
                                             int kt, int lane) {
    const int half = (lane >> 4) & 1;
    Frag16 f;
    f.q[0] = *(const uint4*)(rowPtr + kt + half * 8);
    f.q[1] = *(const uint4*)(rowPtr + kt + 16 + half * 8);
    return f.v;
}

// B-matrix fragment from weight matrix stored N x K row-major.
// lane holds column N = (lane&15); lanes 0-15 -> K=kt..kt+7,kt+16..kt+23; 16-31 -> kt+8..kt+15,kt+24..kt+31
__device__ __forceinline__ v16bf load_b_frag(const unsigned short* W,
                                             int rowStride, int nrow,
                                             int kt, int lane) {
    const int koff = kt + ((lane < 16) ? 0 : 8);
    const unsigned short* p = W + (long)nrow * rowStride + koff;
    Frag16 f;
    f.q[0] = *(const uint4*)(p);
    f.q[1] = *(const uint4*)(p + 16);
    return f.v;
}

__device__ __forceinline__ float wave_sum(float v) {
#pragma unroll
    for (int off = 16; off > 0; off >>= 1) v += __shfl_xor(v, off, 32);
    return v;
}

__device__ __forceinline__ float sigmoidf_(float x) {
    return 1.f / (1.f + expf(-x));
}

// ---------------------------------------------------------------------------
// Problem dims
// ---------------------------------------------------------------------------
#define BZ   32
#define TN   128
#define EN   256
#define HN   256
#define H4   1024
#define H2   512
#define CN   4096
#define LN   129
#define LP   132      // padded L for LDS arrays
#define NEGC (-1000000000.0f)

// ---------------------------------------------------------------------------
// Kernel 0: f32 -> bf16 conversion
// ---------------------------------------------------------------------------
__global__ void k_cvt_bf16(const float* __restrict__ src,
                           unsigned short* __restrict__ dst, int n) {
    int i = blockIdx.x * blockDim.x + threadIdx.x;
    if (i < n) dst[i] = f32_to_bf16_rne(src[i]);
}

// ---------------------------------------------------------------------------
// Kernel A: Xg[bt, 4H] = embed[x[bt]] @ W_ih^T + b_ih + b_hh
// Wave computes 16(M) x 64(N); M=4096, N=1024, K=256
// ---------------------------------------------------------------------------
__global__ __launch_bounds__(128) void kA_xgates(
    const int* __restrict__ x,
    const unsigned short* __restrict__ emb_bf,   // (4096, 256) bf16
    const unsigned short* __restrict__ wih_bf,   // (1024, 256) bf16
    const float* __restrict__ b_ih,
    const float* __restrict__ b_hh,
    float* __restrict__ Xg)                      // (4096, 1024) f32
{
    const int lane  = threadIdx.x & 31;
    const int wid   = threadIdx.x >> 5;
    const int gid   = blockIdx.x * 4 + wid;      // 0..4095
    const int mtile = gid >> 4;                  // 0..255
    const int ng    = gid & 15;                  // 0..15 (64 cols each)

    const int row   = mtile * 16 + (lane & 15);
    const int token = x[row];
    const unsigned short* arow = emb_bf + (long)token * EN;

    v8f acc[4];
#pragma unroll
    for (int s = 0; s < 4; s++) acc[s] = zero8();

    for (int kt = 0; kt < EN; kt += 32) {
        v16bf a = load_a_frag(arow, kt, lane);
#pragma unroll
        for (int s = 0; s < 4; s++) {
            int nrow = ng * 64 + s * 16 + (lane & 15);
            v16bf b  = load_b_frag(wih_bf, EN, nrow, kt, lane);
            acc[s] = wmma_bf16(a, b, acc[s]);
        }
    }
#pragma unroll
    for (int s = 0; s < 4; s++) {
#pragma unroll
        for (int r = 0; r < 8; r++) {
            int m = mtile * 16 + ((lane < 16) ? r : 8 + r);
            int n = ng * 64 + s * 16 + (lane & 15);
            Xg[(long)m * H4 + n] = acc[s][r] + b_ih[n] + b_hh[n];
        }
    }
}

// ---------------------------------------------------------------------------
// Kernel B: persistent 128-step LSTM + sparse attention scan.
// One workgroup, 32 waves. Wave w = (mt = w&1, ht = w>>1) owns a 16x16
// (batch x hidden) patch and computes its i/f/g/o gate tiles so the cell
// update stays in WMMA accumulator registers. Cell state c: 8 VGPRs/thread.
// ---------------------------------------------------------------------------
__global__ __launch_bounds__(1024) void kB_scan(
    const float* __restrict__ Xg,                // (4096, 1024)
    const unsigned short* __restrict__ whh,      // (1024, 256) bf16
    const float* __restrict__ w_t,               // (512)
    float* __restrict__ buf,                     // ws (32, 129, 256) f32
    unsigned short* __restrict__ outs_bf)        // ws (4096, 512) bf16
{
    __shared__ __align__(16) unsigned short hbf[BZ * HN];   // h as bf16 (A matrix)
    __shared__ float hnew[BZ * HN];
    __shared__ float sarr[BZ * LP];    // cached tanh(buf_l) . w2
    __shared__ float logits[BZ * LP];  // logits, then attention weights
    __shared__ float mlog[BZ * LP];    // scratch for top-k selection

    const int tid  = threadIdx.x;
    const int lane = tid & 31;
    const int wid  = tid >> 5;     // 0..31
    const int mt   = wid & 1;      // batch tile
    const int ht   = wid >> 1;     // hidden tile 0..15
    const int bb   = wid;          // attention: one wave per batch row

    // ---- init ----
    for (int i = tid; i < BZ * HN; i += 1024) hbf[i] = 0;
    for (int i = tid; i < BZ * HN; i += 1024) {
        int b = i >> 8, h = i & 255;
        buf[((long)b * LN + 0) * HN + h] = 0.f;     // position 0 stays zero
    }
    if (tid < BZ) sarr[tid * LP + 0] = 0.f;
    float cst[8];
#pragma unroll
    for (int r = 0; r < 8; r++) cst[r] = 0.f;
    __syncthreads();

    for (int t = 0; t < TN; t++) {
        const int rem = t + 1;

        // ---- phase 1: gates = Xg + h @ W_hh^T ; LSTM cell in registers ----
        v8f acc[4];
#pragma unroll
        for (int g = 0; g < 4; g++) acc[g] = zero8();

        const unsigned short* arow = hbf + (mt * 16 + (lane & 15)) * HN;
        for (int kt = 0; kt < HN; kt += 32) {
            v16bf a = load_a_frag(arow, kt, lane);
#pragma unroll
            for (int g = 0; g < 4; g++) {
                int nrow = g * HN + ht * 16 + (lane & 15);
                v16bf b  = load_b_frag(whh, HN, nrow, kt, lane);
                acc[g] = wmma_bf16(a, b, acc[g]);
            }
        }
#pragma unroll
        for (int r = 0; r < 8; r++) {
            int b = mt * 16 + ((lane < 16) ? r : 8 + r);
            int h = ht * 16 + (lane & 15);
            long xb = ((long)b * TN + t) * H4;
            float ig = acc[0][r] + Xg[xb + 0 * HN + h];
            float fg = acc[1][r] + Xg[xb + 1 * HN + h];
            float gg = acc[2][r] + Xg[xb + 2 * HN + h];
            float og = acc[3][r] + Xg[xb + 3 * HN + h];
            float cn = sigmoidf_(fg) * cst[r] + sigmoidf_(ig) * tanhf(gg);
            float hn = sigmoidf_(og) * tanhf(cn);
            cst[r] = cn;
            hnew[b * HN + h] = hn;
        }
        __syncthreads();

        // ---- phase 2: attention, one wave per batch row bb ----
        float hsum = 0.f;
#pragma unroll
        for (int j = 0; j < 8; j++) {
            int h = lane + j * 32;
            hsum += tanhf(hnew[bb * HN + h]) * w_t[h];
        }
        hsum = wave_sum(hsum);   // tanh(h_new).w1 (uniform per wave)

        for (int l = lane; l < rem; l += 32) {
            float lg = hsum + sarr[bb * LP + l];
            logits[bb * LP + l] = lg;
            mlog[bb * LP + l]   = lg;
        }

        if (rem > 5) {                       // top-(K+1) sparse softmax-ish
            float kept = NEGC;
            for (int it = 0; it < 6; it++) {
                float mv = -3.0e38f; int mi = 0x7fffffff;
                for (int l = lane; l < rem; l += 32) {
                    float v = mlog[bb * LP + l];
                    if (v > mv) { mv = v; mi = l; }
                }
#pragma unroll
                for (int off = 16; off > 0; off >>= 1) {
                    float ov = __shfl_xor(mv, off, 32);
                    int   oi = __shfl_xor(mi, off, 32);
                    if (ov > mv || (ov == mv && oi < mi)) { mv = ov; mi = oi; }
                }
                kept = mv;
                if (lane == 0 && mi < LP) mlog[bb * LP + mi] = -3.0e38f;
            }
            float delta = kept - 1e-7f;
            float wsum = 0.f;
            for (int l = lane; l < rem; l += 32) {
                float w = logits[bb * LP + l] - delta;
                w = (w > 0.f) ? w : 0.f;
                logits[bb * LP + l] = w;
                wsum += w;
            }
            wsum = wave_sum(wsum);
            float inv = 1.f / (wsum + 1e-7f);
            for (int l = lane; l < rem; l += 32) logits[bb * LP + l] *= inv;
        }
        // else: weights = raw valid logits (wd path) — already in logits[]

        // ---- attn_c = sum_l w_l * buf[b,l,:]  (<=6 nonzero when rem>5) ----
        float ac[8];
#pragma unroll
        for (int j = 0; j < 8; j++) ac[j] = 0.f;
        for (int l = 0; l < rem; l++) {
            float w = logits[bb * LP + l];       // wave-uniform
            if (w != 0.f) {
                const float* bp = buf + ((long)bb * LN + l) * HN;
#pragma unroll
                for (int j = 0; j < 8; j++) ac[j] += w * bp[lane + j * 32];
            }
        }

        // ---- finalize: h_out = h_new + attn_c; write buf/outs/h_bf16/s ----
        float ssum = 0.f;
        float* bout = buf + ((long)bb * LN + rem) * HN;
        unsigned short* obf = outs_bf + ((long)bb * TN + t) * H2;
#pragma unroll
        for (int j = 0; j < 8; j++) {
            int h = lane + j * 32;
            float ho = hnew[bb * HN + h] + ac[j];
            bout[h]        = ho;
            obf[h]         = f32_to_bf16_rne(ho);      // hs
            obf[HN + h]    = f32_to_bf16_rne(ac[j]);   // acs
            hbf[bb * HN + h] = f32_to_bf16_rne(ho);    // next-step A matrix
            ssum += tanhf(ho) * w_t[HN + h];
        }
        ssum = wave_sum(ssum);
        if (lane == 0) sarr[bb * LP + rem] = ssum;     // cache for future steps
        __syncthreads();
    }
}

// ---------------------------------------------------------------------------
// Kernel C: out[bt, c] = outs[bt, :] . fc_W[c, :] + fc_b[c]
// M=4096, N=4096, K=512. Wave computes 16 x 64.
// ---------------------------------------------------------------------------
__global__ __launch_bounds__(128) void kC_fc(
    const unsigned short* __restrict__ outs_bf,  // (4096, 512) bf16
    const unsigned short* __restrict__ fcw_bf,   // (4096, 512) bf16
    const float* __restrict__ fc_b,
    float* __restrict__ out)                     // (4096, 4096) f32
{
    const int lane  = threadIdx.x & 31;
    const int wid   = threadIdx.x >> 5;
    const int gid   = blockIdx.x * 4 + wid;      // 0..16383
    const int mtile = gid >> 6;                  // 0..255
    const int ng    = gid & 63;                  // 0..63

    const unsigned short* arow =
        outs_bf + (long)(mtile * 16 + (lane & 15)) * H2;

    v8f acc[4];
#pragma unroll
    for (int s = 0; s < 4; s++) acc[s] = zero8();

    for (int kt = 0; kt < H2; kt += 32) {
        v16bf a = load_a_frag(arow, kt, lane);
#pragma unroll
        for (int s = 0; s < 4; s++) {
            int n = ng * 64 + s * 16 + (lane & 15);
            v16bf b = load_b_frag(fcw_bf, H2, n, kt, lane);
            acc[s] = wmma_bf16(a, b, acc[s]);
        }
    }
#pragma unroll
    for (int s = 0; s < 4; s++) {
#pragma unroll
        for (int r = 0; r < 8; r++) {
            int m = mtile * 16 + ((lane < 16) ? r : 8 + r);
            int n = ng * 64 + s * 16 + (lane & 15);
            out[(long)m * CN + n] = acc[s][r] + fc_b[n];
        }
    }
}

// ---------------------------------------------------------------------------
// Host launcher
// ---------------------------------------------------------------------------
extern "C" void kernel_launch(void* const* d_in, const int* in_sizes, int n_in,
                              void* d_out, int out_size, void* d_ws, size_t ws_size,
                              hipStream_t stream) {
    (void)in_sizes; (void)n_in; (void)out_size; (void)ws_size;

    const int*   x      = (const int*)  d_in[0];
    const float* embedW = (const float*)d_in[1];
    const float* W_ih   = (const float*)d_in[2];
    const float* W_hh   = (const float*)d_in[3];
    const float* b_ih   = (const float*)d_in[4];
    const float* b_hh   = (const float*)d_in[5];
    const float* w_t    = (const float*)d_in[6];
    const float* fc_W   = (const float*)d_in[7];
    const float* fc_b   = (const float*)d_in[8];
    float*       out    = (float*)d_out;

    // workspace layout (all offsets 256B-aligned)
    char* ws = (char*)d_ws;
    size_t off = 0;
    float* Xg = (float*)(ws + off);            off += (size_t)4096 * H4 * 4;   // 16 MB
    float* buf = (float*)(ws + off);           off += (size_t)BZ * LN * HN * 4;// ~4.03 MB
    unsigned short* outs_bf = (unsigned short*)(ws + off); off += (size_t)4096 * H2 * 2; // 4 MB
    unsigned short* emb_bf  = (unsigned short*)(ws + off); off += (size_t)CN * EN * 2;   // 2 MB
    unsigned short* wih_bf  = (unsigned short*)(ws + off); off += (size_t)H4 * EN * 2;   // 0.5 MB
    unsigned short* whh_bf  = (unsigned short*)(ws + off); off += (size_t)H4 * HN * 2;   // 0.5 MB
    unsigned short* fcw_bf  = (unsigned short*)(ws + off); off += (size_t)CN * H2 * 2;   // 4 MB

    // 1) weight conversions to bf16
    {
        int n;
        n = CN * EN; k_cvt_bf16<<<(n + 255) / 256, 256, 0, stream>>>(embedW, emb_bf, n);
        n = H4 * EN; k_cvt_bf16<<<(n + 255) / 256, 256, 0, stream>>>(W_ih,   wih_bf, n);
        n = H4 * HN; k_cvt_bf16<<<(n + 255) / 256, 256, 0, stream>>>(W_hh,   whh_bf, n);
        n = CN * H2; k_cvt_bf16<<<(n + 255) / 256, 256, 0, stream>>>(fc_W,   fcw_bf, n);
    }

    // 2) input-side gate pre-compute (embedding gather + GEMM + biases)
    kA_xgates<<<1024, 128, 0, stream>>>(x, emb_bf, wih_bf, b_ih, b_hh, Xg);

    // 3) sequential LSTM + sparse attention scan (persistent workgroup)
    kB_scan<<<1, 1024, 0, stream>>>(Xg, whh_bf, w_t, buf, outs_bf);

    // 4) final FC GEMM
    kC_fc<<<4096, 128, 0, stream>>>(outs_bf, fcw_bf, fc_b, out);
}
